// PanelSegRetinaNet_69672959475730
// MI455X (gfx1250) — matrix operand order repaired and checked by reading, not compiled, not run on hardware
//
#include <hip/hip_runtime.h>
#include <hip/hip_bf16.h>

// ---------------------------------------------------------------------------
// Types for CDNA5 WMMA (wave32): D(16x16 f32) = A(16x32 bf16) x B(32x16 bf16) + C
// ---------------------------------------------------------------------------
typedef __attribute__((ext_vector_type(16))) __bf16 v16bf;
typedef __attribute__((ext_vector_type(8)))  float  v8f;

#define CIN 256
#define TOTA 120087      // total anchors per batch = 9 * (10000+2500+625+169+49)
#define NCAND 4441       // 1000+1000+1000+1000+441
#define IMGSZ 800.0f
#define SCORE_TH 0.05f
#define NMS_TH 0.5f
#define SCALE_CLAMP 4.135166556742356f  // log(1000/16)

__device__ __forceinline__ unsigned short f2bf(float f) {
  unsigned int u = __float_as_uint(f);
  unsigned int r = u + 0x7FFFu + ((u >> 16) & 1u);   // round-to-nearest-even
  return (unsigned short)(r >> 16);
}
__device__ __forceinline__ float bf2f(unsigned short h) {
  return __uint_as_float(((unsigned int)h) << 16);
}

// ---------------------------------------------------------------------------
// NCHW f32 -> NHWC bf16
// ---------------------------------------------------------------------------
__global__ void nchw_to_nhwc_bf16(const float* __restrict__ in,
                                  unsigned short* __restrict__ out,
                                  int H, int W) {
  long long idx = (long long)blockIdx.x * blockDim.x + threadIdx.x;
  long long total = 2LL * CIN * H * W;
  if (idx >= total) return;
  int c = (int)(idx % CIN);
  int p = (int)((idx / CIN) % ((long long)H * W));
  int n = (int)(idx / ((long long)CIN * H * W));
  int y = p / W, x = p % W;
  out[idx] = f2bf(in[(((long long)n * CIN + c) * H + y) * W + x]);
}

// Weights [Cout=256][Cin=256][3][3] f32 -> [tap=9][Cout][Cin] bf16
__global__ void repack_weights_bf16(const float* __restrict__ w,
                                    unsigned short* __restrict__ o) {
  int idx = blockIdx.x * blockDim.x + threadIdx.x;
  if (idx >= 9 * 256 * 256) return;
  int ci = idx % 256;
  int co = (idx / 256) % 256;
  int t  = idx / (256 * 256);
  o[((size_t)t * 256 + co) * 256 + ci] = f2bf(w[((size_t)(co * 256 + ci)) * 9 + t]);
}

// ---------------------------------------------------------------------------
// 3x3 conv, Cin=Cout=256, NHWC bf16, implicit GEMM via v_wmma_f32_16x16x32_bf16.
// Block: 256 threads = 8 waves, block tile = 128 couts x 32 pixels (one row).
// Each wave: 16 couts x 32 pixels (2 accumulators sharing the A fragment).
// Halo tile staged to LDS with GLOBAL_LOAD_ASYNC_TO_LDS_B128 (ASYNCcnt).
// grid = ( ceil(W/32), H, N*2 )
// ---------------------------------------------------------------------------
__global__ __launch_bounds__(256)
void conv3x3_wmma_bf16(const unsigned short* __restrict__ in,   // [N][H][W][256]
                       const unsigned short* __restrict__ wt,   // [9][256][256]
                       const float* __restrict__ bias,          // [256]
                       unsigned short* __restrict__ out,        // [N][H][W][256]
                       int H, int W, int relu) {
  __shared__ unsigned short smem[3 * 34 * 32];  // [row 3][col 34][cin 32] halo tile

  const int tid  = threadIdx.x;
  const int lane = tid & 31;
  const int wv   = tid >> 5;      // wave 0..7 -> cout sub-tile
  const int x0   = blockIdx.x * 32;
  const int y0   = blockIdx.y;
  const int n    = blockIdx.z >> 1;
  const int m0   = (blockIdx.z & 1) * 128 + wv * 16;  // cout base of this wave

  const int half = lane >> 4;     // 0: lanes 0-15, 1: lanes 16-31
  const int l16  = lane & 15;

  // Accumulators (C/D layout: VGPR r -> M = r + 8*half, N = lane%16)
  v8f acc0, acc1;
#pragma unroll
  for (int r = 0; r < 8; ++r) { acc0[r] = bias[m0 + half * 8 + r]; acc1[r] = acc0[r]; }

  const size_t planeIn = (size_t)n * H * W * CIN;

  // A-fragment addressing (16-bit A 16x32 layout): two contiguous 16B chunks
  const int aK0   = half * 16;        // halves 0..7  (lanes0-15) / 16..23 (lanes16-31)
  const int aK1   = half * 16 + 8;    // halves 8..15 / 24..31
  const int acout = m0 + l16;
  // B-fragment addressing (16-bit B 32x16): 16 contiguous K halves per lane
  const int bK = half * 16;           // halves offset within the 32-cin chunk

  for (int cb = 0; cb < 8; ++cb) {    // 8 chunks of 32 input channels
    __syncthreads();
    // Stage halo tile: 102 (row,col) sites x 32 cin halves = 408 x 16B chunks.
    // In-bounds chunks go memory->LDS via the CDNA5 async tensor path.
    for (int q = tid; q < 408; q += 256) {
      int site = q >> 2, part = q & 3;
      int r = site / 34, c = site - r * 34;
      int gy = y0 - 1 + r, gx = x0 - 1 + c;
      unsigned short* dst = &smem[site * 32 + part * 8];
      if (gy >= 0 && gy < H && gx >= 0 && gx < W) {
        unsigned long long gaddr = (unsigned long long)(
            in + planeIn + ((size_t)gy * W + gx) * CIN + cb * 32 + part * 8);
        unsigned int ldsOff = (unsigned int)(unsigned long long)dst;  // AS3 offset
        asm volatile("global_load_async_to_lds_b128 %0, %1, off"
                     :: "v"(ldsOff), "v"(gaddr) : "memory");
      } else {
        *reinterpret_cast<uint4*>(dst) = make_uint4(0u, 0u, 0u, 0u);
      }
    }
    asm volatile("s_wait_asynccnt 0x0" ::: "memory");
    __syncthreads();

#pragma unroll
    for (int t = 0; t < 9; ++t) {
      const int ky = t / 3, kx = t - ky * 3;
      union { uint4 q[2]; v16bf v; } Af;
      const unsigned short* wp = wt + ((size_t)t * 256 + acout) * CIN + cb * 32;
      Af.q[0] = *reinterpret_cast<const uint4*>(wp + aK0);
      Af.q[1] = *reinterpret_cast<const uint4*>(wp + aK1);

      // B fragments for pixel tiles 0 (cols l16+kx) and 1 (cols 16+l16+kx)
      union { uint4 q[2]; v16bf v; } B0, B1;
      const unsigned short* sp0 = &smem[((ky * 34) + l16 + kx) * 32 + bK];
      B0.q[0] = *reinterpret_cast<const uint4*>(sp0);
      B0.q[1] = *reinterpret_cast<const uint4*>(sp0 + 8);
      const unsigned short* sp1 = sp0 + 16 * 32;
      B1.q[0] = *reinterpret_cast<const uint4*>(sp1);
      B1.q[1] = *reinterpret_cast<const uint4*>(sp1 + 8);

      acc0 = __builtin_amdgcn_wmma_f32_16x16x32_bf16(
          false, Af.v, false, B0.v, (short)0, acc0, false, false);
      acc1 = __builtin_amdgcn_wmma_f32_16x16x32_bf16(
          false, Af.v, false, B1.v, (short)0, acc1, false, false);
    }
  }

  // Epilogue: ReLU, cvt to bf16, 16B store of couts [m0+8*half .. +7] per pixel
  const size_t outBase = planeIn + (size_t)y0 * W * CIN + m0 + half * 8;
#pragma unroll
  for (int ptile = 0; ptile < 2; ++ptile) {
    const int px = x0 + ptile * 16 + l16;
    if (px < W) {
      v8f* accp = ptile ? &acc1 : &acc0;
      unsigned int packed[4];
#pragma unroll
      for (int r = 0; r < 4; ++r) {
        float lo = (*accp)[2 * r], hi = (*accp)[2 * r + 1];
        if (relu) { lo = fmaxf(lo, 0.f); hi = fmaxf(hi, 0.f); }
        packed[r] = (unsigned int)f2bf(lo) | ((unsigned int)f2bf(hi) << 16);
      }
      uint4 v; v.x = packed[0]; v.y = packed[1]; v.z = packed[2]; v.w = packed[3];
      *reinterpret_cast<uint4*>(out + outBase + (size_t)px * CIN) = v;
    }
  }
}

// ---------------------------------------------------------------------------
// Final head convs (Cout = 9 scores w/ sigmoid, or 36 deltas). Direct VALU.
// out[n*batchStride + pix*COUT + co]  (base pointer pre-offset per level)
// ---------------------------------------------------------------------------
template <int COUT, bool SIG>
__global__ void head_conv(const unsigned short* __restrict__ in,  // NHWC bf16
                          const float* __restrict__ w,            // [COUT][256][3][3]
                          const float* __restrict__ b,
                          float* __restrict__ out,
                          int H, int W, int batchStride) {
  long long idx = (long long)blockIdx.x * blockDim.x + threadIdx.x;
  long long total = 2LL * H * W * COUT;
  if (idx >= total) return;
  int co  = (int)(idx % COUT);
  int pix = (int)((idx / COUT) % ((long long)H * W));
  int n   = (int)(idx / ((long long)COUT * H * W));
  int y = pix / W, x = pix % W;
  float acc = b[co];
  for (int ky = 0; ky < 3; ++ky) {
    int gy = y + ky - 1; if (gy < 0 || gy >= H) continue;
    for (int kx = 0; kx < 3; ++kx) {
      int gx = x + kx - 1; if (gx < 0 || gx >= W) continue;
      const unsigned short* ip = in + (((size_t)n * H + gy) * W + gx) * CIN;
      const float* wp = w + (size_t)co * CIN * 9 + ky * 3 + kx;
      for (int ci = 0; ci < CIN; ++ci) acc += bf2f(ip[ci]) * wp[ci * 9];
    }
  }
  if (SIG) acc = 1.0f / (1.0f + __expf(-acc));
  out[(size_t)n * batchStride + (size_t)pix * COUT + co] = acc;
}

// ---------------------------------------------------------------------------
// Per (level,batch) top-k by iterative block argmax (mutates scores copy)
// ---------------------------------------------------------------------------
__global__ __launch_bounds__(256)
void topk_kernel(float* __restrict__ scores, float* __restrict__ topv,
                 int* __restrict__ topi) {
  const int LN[5]   = {90000, 22500, 5625, 1521, 441};
  const int AOFF[5] = {0, 90000, 112500, 118125, 119646};
  const int KL[5]   = {1000, 1000, 1000, 1000, 441};
  int l = blockIdx.x, n = blockIdx.y;
  float* s  = scores + (size_t)n * TOTA + AOFF[l];
  float* tv = topv + (n * 5 + l) * 1000;
  int*   ti = topi + (n * 5 + l) * 1000;
  int nE = LN[l], k = KL[l];
  __shared__ float sv[256];
  __shared__ int   si[256];
  for (int t = 0; t < k; ++t) {
    float bv = -__builtin_inff(); int bi = 0;
    for (int e = threadIdx.x; e < nE; e += 256) {
      float v = s[e];
      if (v > bv) { bv = v; bi = e; }
    }
    sv[threadIdx.x] = bv; si[threadIdx.x] = bi;
    __syncthreads();
    for (int st = 128; st > 0; st >>= 1) {
      if (threadIdx.x < st && sv[threadIdx.x + st] > sv[threadIdx.x]) {
        sv[threadIdx.x] = sv[threadIdx.x + st];
        si[threadIdx.x] = si[threadIdx.x + st];
      }
      __syncthreads();
    }
    if (threadIdx.x == 0) {
      tv[t] = sv[0]; ti[t] = si[0];
      s[si[0]] = -__builtin_inff();
    }
    __threadfence_block();
    __syncthreads();
  }
}

// ---------------------------------------------------------------------------
// Decode: anchors computed in-kernel; box regression; clamp to [0, 800]
// ---------------------------------------------------------------------------
__global__ void decode_kernel(const float* __restrict__ topv,
                              const int* __restrict__ topi,
                              const float* __restrict__ deltas,
                              float* __restrict__ candBox,
                              float* __restrict__ candKey) {
  const int LW[5]     = {100, 50, 25, 13, 7};
  const int STRIDE[5] = {8, 16, 32, 64, 128};
  const int SIZES[5]  = {32, 64, 128, 256, 512};
  const int AOFF[5]   = {0, 90000, 112500, 118125, 119646};
  const int KL[5]     = {1000, 1000, 1000, 1000, 441};
  const int COFF[5]   = {0, 1000, 2000, 3000, 4000};
  int gid = blockIdx.x * 256 + threadIdx.x;
  if (gid >= 10000) return;
  int n = gid / 5000, rem = gid % 5000;
  int l = rem / 1000, t = rem % 1000;
  if (t >= KL[l]) return;
  int slot = COFF[l] + t;
  int idx  = topi[(n * 5 + l) * 1000 + t];
  float sc = topv[(n * 5 + l) * 1000 + t];
  int a = idx % 9, pix = idx / 9;
  int W = LW[l];
  int x = pix % W, y = pix / W;
  float scale = exp2f((float)(a / 3) * (1.0f / 3.0f));
  int aridx = a % 3;
  float ar = (aridx == 0) ? 0.5f : ((aridx == 1) ? 1.0f : 2.0f);
  float size = (float)SIZES[l];
  float area = (size * scale) * (size * scale);
  float wA = sqrtf(area / ar);
  float hA = ar * wA;
  float acx = (float)x * (float)STRIDE[l];
  float acy = (float)y * (float)STRIDE[l];
  const float* d = deltas + ((size_t)n * TOTA + AOFF[l] + idx) * 4;
  float dw = fminf(d[2], SCALE_CLAMP), dh = fminf(d[3], SCALE_CLAMP);
  float pcx = d[0] * wA + acx, pcy = d[1] * hA + acy;
  float pw = __expf(dw) * wA, ph = __expf(dh) * hA;
  float* cb = candBox + ((size_t)n * NCAND + slot) * 4;
  cb[0] = fminf(fmaxf(pcx - 0.5f * pw, 0.f), IMGSZ);
  cb[1] = fminf(fmaxf(pcy - 0.5f * ph, 0.f), IMGSZ);
  cb[2] = fminf(fmaxf(pcx + 0.5f * pw, 0.f), IMGSZ);
  cb[3] = fminf(fmaxf(pcy + 0.5f * ph, 0.f), IMGSZ);
  candKey[(size_t)n * NCAND + slot] = (sc > SCORE_TH) ? sc : -__builtin_inff();
}

// ---------------------------------------------------------------------------
// Per-batch: 5-way merge of sorted per-level lists, greedy NMS, emit 100x6
// ---------------------------------------------------------------------------
__global__ __launch_bounds__(256)
void nms_out_kernel(const float* __restrict__ candBox,
                    const float* __restrict__ candKey,
                    float* __restrict__ out) {
  int n = blockIdx.x;
  const float* box = candBox + (size_t)n * NCAND * 4;
  const float* key = candKey + (size_t)n * NCAND;
  __shared__ int order[NCAND];
  __shared__ unsigned char keep[NCAND];
  if (threadIdx.x == 0) {
    const int COFF[5] = {0, 1000, 2000, 3000, 4000};
    const int KL[5]   = {1000, 1000, 1000, 1000, 441};
    int h[5] = {0, 0, 0, 0, 0};
    for (int m = 0; m < NCAND; ++m) {
      int best = -1; float bk = 0.f;
      for (int l = 0; l < 5; ++l) {
        if (h[l] < KL[l]) {
          float kk = key[COFF[l] + h[l]];
          if (best < 0 || kk > bk) { bk = kk; best = l; }
        }
      }
      order[m] = COFF[best] + h[best];
      h[best]++;
    }
  }
  __syncthreads();
  for (int m = threadIdx.x; m < NCAND; m += 256)
    keep[m] = (key[order[m]] > 0.0f) ? 1 : 0;   // valid = score > threshold
  __syncthreads();
  for (int i = 0; i < NCAND; ++i) {
    __syncthreads();
    if (!keep[i]) continue;                      // uniform across block
    int oi = order[i];
    float x1 = box[oi * 4 + 0], y1 = box[oi * 4 + 1];
    float x2 = box[oi * 4 + 2], y2 = box[oi * 4 + 3];
    float ai = fmaxf(x2 - x1, 0.f) * fmaxf(y2 - y1, 0.f);
    for (int j = i + 1 + threadIdx.x; j < NCAND; j += 256) {
      if (!keep[j]) continue;
      int oj = order[j];
      float u1 = box[oj * 4 + 0], v1 = box[oj * 4 + 1];
      float u2 = box[oj * 4 + 2], v2 = box[oj * 4 + 3];
      float iw = fmaxf(fminf(x2, u2) - fmaxf(x1, u1), 0.f);
      float ih = fmaxf(fminf(y2, v2) - fmaxf(y1, v1), 0.f);
      float inter = iw * ih;
      float uni = ai + fmaxf(u2 - u1, 0.f) * fmaxf(v2 - v1, 0.f) - inter;
      float iou = (uni > 0.f) ? inter / uni : 0.f;
      if (iou > NMS_TH) keep[j] = 0;
    }
  }
  __syncthreads();
  float* o = out + n * 600;
  for (int m = threadIdx.x; m < 600; m += 256) o[m] = 0.f;
  __syncthreads();
  if (threadIdx.x == 0) {
    int cnt = 0;
    for (int m = 0; m < NCAND && cnt < 100; ++m) {
      if (keep[m]) {
        int oi = order[m];
        o[cnt * 6 + 0] = box[oi * 4 + 0];
        o[cnt * 6 + 1] = box[oi * 4 + 1];
        o[cnt * 6 + 2] = box[oi * 4 + 2];
        o[cnt * 6 + 3] = box[oi * 4 + 3];
        o[cnt * 6 + 4] = key[oi];
        o[cnt * 6 + 5] = 0.f;                   // single class
        cnt++;
      }
    }
  }
}

// ---------------------------------------------------------------------------
// Host orchestration (all launches on `stream`; workspace-only scratch)
// ---------------------------------------------------------------------------
extern "C" void kernel_launch(void* const* d_in, const int* in_sizes, int n_in,
                              void* d_out, int out_size, void* d_ws, size_t ws_size,
                              hipStream_t stream) {
  (void)in_sizes; (void)n_in; (void)out_size; (void)ws_size;
  const float* feat[5] = {(const float*)d_in[0], (const float*)d_in[1],
                          (const float*)d_in[2], (const float*)d_in[3],
                          (const float*)d_in[4]};
  const float* cls_w   = (const float*)d_in[5];
  const float* cls_b   = (const float*)d_in[6];
  const float* box_w   = (const float*)d_in[7];
  const float* box_b   = (const float*)d_in[8];
  const float* score_w = (const float*)d_in[9];
  const float* score_b = (const float*)d_in[10];
  const float* pred_w  = (const float*)d_in[11];
  const float* pred_b  = (const float*)d_in[12];

  const int LH[5] = {100, 50, 25, 13, 7};
  const int LW[5] = {100, 50, 25, 13, 7};
  const int AOFF[5] = {0, 90000, 112500, 118125, 119646};

  char* ws = (char*)d_ws;
  size_t off = 0;
  auto alloc = [&](size_t bytes) {
    size_t o = off;
    off += (bytes + 255) & ~(size_t)255;
    return o;
  };
  size_t featOff[5];
  for (int l = 0; l < 5; ++l)
    featOff[l] = alloc((size_t)2 * LH[l] * LW[l] * CIN * 2);
  size_t pingA   = alloc((size_t)2 * 100 * 100 * CIN * 2);
  size_t pingB   = alloc((size_t)2 * 100 * 100 * CIN * 2);
  size_t wclsOff = alloc((size_t)4 * 9 * 256 * 256 * 2);
  size_t wboxOff = alloc((size_t)4 * 9 * 256 * 256 * 2);
  size_t scoresOff  = alloc((size_t)2 * TOTA * 4);
  size_t deltasOff  = alloc((size_t)2 * TOTA * 4 * 4);
  size_t topvOff    = alloc((size_t)2 * 5 * 1000 * 4);
  size_t topiOff    = alloc((size_t)2 * 5 * 1000 * 4);
  size_t candBoxOff = alloc((size_t)2 * NCAND * 4 * 4);
  size_t candKeyOff = alloc((size_t)2 * NCAND * 4);

  // 1. Convert features to NHWC bf16
  for (int l = 0; l < 5; ++l) {
    long long total = 2LL * CIN * LH[l] * LW[l];
    int blocks = (int)((total + 255) / 256);
    nchw_to_nhwc_bf16<<<blocks, 256, 0, stream>>>(
        feat[l], (unsigned short*)(ws + featOff[l]), LH[l], LW[l]);
  }
  // 2. Repack subnet weights to bf16 [tap][Cout][Cin]
  for (int i = 0; i < 4; ++i) {
    int blocks = (9 * 256 * 256 + 255) / 256;
    repack_weights_bf16<<<blocks, 256, 0, stream>>>(
        cls_w + (size_t)i * 256 * 256 * 9,
        (unsigned short*)(ws + wclsOff) + (size_t)i * 9 * 256 * 256);
    repack_weights_bf16<<<blocks, 256, 0, stream>>>(
        box_w + (size_t)i * 256 * 256 * 9,
        (unsigned short*)(ws + wboxOff) + (size_t)i * 9 * 256 * 256);
  }

  // 3. Per level: WMMA subnet chains + head convs
  for (int l = 0; l < 5; ++l) {
    int H = LH[l], W = LW[l];
    dim3 grid((W + 31) / 32, H, 2 * 2);   // z: n*2 + coutBlk(0..1)
    unsigned short* fin = (unsigned short*)(ws + featOff[l]);
    unsigned short* A = (unsigned short*)(ws + pingA);
    unsigned short* B = (unsigned short*)(ws + pingB);
    const unsigned short* wc = (const unsigned short*)(ws + wclsOff);
    const unsigned short* wb = (const unsigned short*)(ws + wboxOff);
    const size_t WSTEP = (size_t)9 * 256 * 256;

    // cls subnet: fin -> A -> B -> A -> B, then score head (sigmoid)
    conv3x3_wmma_bf16<<<grid, 256, 0, stream>>>(fin, wc + 0 * WSTEP, cls_b + 0,   A, H, W, 1);
    conv3x3_wmma_bf16<<<grid, 256, 0, stream>>>(A,   wc + 1 * WSTEP, cls_b + 256, B, H, W, 1);
    conv3x3_wmma_bf16<<<grid, 256, 0, stream>>>(B,   wc + 2 * WSTEP, cls_b + 512, A, H, W, 1);
    conv3x3_wmma_bf16<<<grid, 256, 0, stream>>>(A,   wc + 3 * WSTEP, cls_b + 768, B, H, W, 1);
    {
      long long total = 2LL * H * W * 9;
      int blocks = (int)((total + 255) / 256);
      head_conv<9, true><<<blocks, 256, 0, stream>>>(
          B, score_w, score_b, (float*)(ws + scoresOff) + AOFF[l], H, W, TOTA);
    }
    // box subnet
    conv3x3_wmma_bf16<<<grid, 256, 0, stream>>>(fin, wb + 0 * WSTEP, box_b + 0,   A, H, W, 1);
    conv3x3_wmma_bf16<<<grid, 256, 0, stream>>>(A,   wb + 1 * WSTEP, box_b + 256, B, H, W, 1);
    conv3x3_wmma_bf16<<<grid, 256, 0, stream>>>(B,   wb + 2 * WSTEP, box_b + 512, A, H, W, 1);
    conv3x3_wmma_bf16<<<grid, 256, 0, stream>>>(A,   wb + 3 * WSTEP, box_b + 768, B, H, W, 1);
    {
      long long total = 2LL * H * W * 36;
      int blocks = (int)((total + 255) / 256);
      head_conv<36, false><<<blocks, 256, 0, stream>>>(
          B, pred_w, pred_b, (float*)(ws + deltasOff) + (size_t)AOFF[l] * 4, H, W, TOTA * 4);
    }
  }

  // 4. Top-k per (level, batch)
  topk_kernel<<<dim3(5, 2), 256, 0, stream>>>(
      (float*)(ws + scoresOff), (float*)(ws + topvOff), (int*)(ws + topiOff));

  // 5. Decode candidates
  decode_kernel<<<(10000 + 255) / 256, 256, 0, stream>>>(
      (const float*)(ws + topvOff), (const int*)(ws + topiOff),
      (const float*)(ws + deltasOff),
      (float*)(ws + candBoxOff), (float*)(ws + candKeyOff));

  // 6. Merge + NMS + output (per batch)
  nms_out_kernel<<<2, 256, 0, stream>>>(
      (const float*)(ws + candBoxOff), (const float*)(ws + candKeyOff),
      (float*)d_out);
}
